// ActiveInference_52458730553348
// MI455X (gfx1250) — matrix operand order repaired and checked by reading, not compiled, not run on hardware
//
#include <hip/hip_runtime.h>

typedef __attribute__((ext_vector_type(16))) __bf16 v16bf;
typedef __attribute__((ext_vector_type(8)))  float  v8f;

#define TT    500
#define NB    64
#define NS    64
#define NA    16
#define DOBS  32
#define DCTL  8
#define HH    30
#define TB    (TT*NB)          // 32000 rows, divisible by 16
static __device__ __host__ constexpr float TWO_PI_F = 6.2831853071795864f;

// ---------------- workspace layout (bytes) ----------------
#define OFF_BP    0u                 // Bp f32 [16][64][64]        = 262144
#define OFF_Q     0x40000u           // Q  f32 [16][64]            = 4096
#define OFF_B0    0x41000u           // b0 f32 [64]
#define OFF_WOBS  0x42000u           // W_obs bf16 fragments 72*512 = 73728 B
#define OFF_WCTL  0x54000u           // W_ctl bf16 fragments 2*512  = 2048 B
#define OFF_LPO   0x60000u           // logp_o f32 [32000][64]     = 8192000
#define OFF_LPU   0x840000u          // logp_u f32 [32000][16]     = 2048000
#define OFF_PA    0xA40000u          // p_a    f32 [32000][16]     = 2048000

// =========================================================
// Per-component MVN parameter kernel: triangular inverse ->
// precision P, q = P mu, const; writes W in WMMA-B fragment order (bf16).
// One wave (32 threads) per mixture component.
// =========================================================
template<int D, int NT, int NCH>
__global__ __launch_bounds__(32)
void mvn_params_kernel(const float* __restrict__ mu,
                       const float* __restrict__ lv,
                       const float* __restrict__ tl,
                       __bf16* __restrict__ Wf)
{
  constexpr int NPAIR = D * (D + 1) / 2;
  __shared__ float L[D][D], Li[D][D], P[D][D], q[D], mus[D];
  __shared__ float cshared;
  const int k = blockIdx.x, t = threadIdx.x;

  for (int idx = t; idx < D * D; idx += 32) {
    int i = idx / D, j = idx % D;
    float v = (i > j) ? tl[((size_t)k * D + i) * D + j]
                      : (i == j ? expf(lv[k * D + i]) : 0.0f);
    L[i][j] = v;
  }
  if (t < D) mus[t] = mu[k * D + t];
  __syncthreads();

  // forward substitution: column t of L^-1
  if (t < D) {
    const int c = t;
    for (int i = 0; i < D; ++i) {
      float v;
      if (i < c)       v = 0.0f;
      else if (i == c) v = 1.0f / L[c][c];
      else {
        float s = 0.0f;
        for (int j2 = c; j2 < i; ++j2) s += L[i][j2] * Li[j2][c];
        v = -s / L[i][i];
      }
      Li[i][c] = v;
    }
  }
  __syncthreads();

  // P = Li^T Li
  for (int idx = t; idx < D * D; idx += 32) {
    int i = idx / D, j = idx % D;
    float s = 0.0f;
    for (int m = 0; m < D; ++m) s += Li[m][i] * Li[m][j];
    P[i][j] = s;
  }
  __syncthreads();
  if (t < D) {
    float s = 0.0f;
    for (int j = 0; j < D; ++j) s += P[t][j] * mus[j];
    q[t] = s;
  }
  __syncthreads();
  if (t == 0) {
    float cq = 0.0f, slv = 0.0f;
    for (int i = 0; i < D; ++i) { cq += mus[i] * q[i]; slv += lv[k * D + i]; }
    cshared = -0.5f * cq - slv - 0.5f * (float)D * logf(TWO_PI_F);
  }
  __syncthreads();

  // W[fi,k] in WMMA B-fragment order:
  // frag = chunk*NT + (k/16); lane = (k%16) + 16*((fi%32)/16); elem = fi%16
  const int col = k & 15, nt = k >> 4;
  for (int fi = t; fi < NCH * 32; fi += 32) {
    float val;
    if (fi < D) val = q[fi];
    else if (fi < D + NPAIR) {
      int rem = fi - D, i = 0, cnt = D;
      while (rem >= cnt) { rem -= cnt; --cnt; ++i; }
      int j = i + rem;
      val = (i == j) ? -0.5f * P[i][i] : -P[i][j];
    } else if (fi == D + NPAIR) val = cshared;
    else val = 0.0f;
    int c = fi >> 5, r = fi & 31, gg = r >> 4, e = r & 15;
    Wf[((size_t)(c * NT + nt) * 32 + (col + 16 * gg)) * 16 + e] = (__bf16)val;
  }
}

// =========================================================
// WMMA GEMM: logp[tb, k] = F(x_tb) . W[:,k]
// Quadratic features F = [x_i, x_i*x_j (i<=j), 1, 0-pad], bf16.
// One wave per 16-row tile. v_wmma_f32_16x16x32_bf16, K-chained.
// =========================================================
template<int D, int NCH, int NT, int NK>
__global__ __launch_bounds__(32)
void gemm_logp_kernel(const float* __restrict__ x,
                      const __bf16* __restrict__ Wf,
                      float* __restrict__ out)
{
  constexpr int NPAIR = D * (D + 1) / 2;
  constexpr int KF = NCH * 32;
  __shared__ float xs[16][D + 2];
  __shared__ unsigned short lut[KF];
  const int lane = threadIdx.x;
  const int row  = lane & 15, g = lane >> 4;
  const long base = (long)blockIdx.x * 16;

  __builtin_prefetch(Wf, 0, 3);

  for (int idx = lane; idx < 16 * D; idx += 32)
    xs[idx / D][idx % D] = x[(base + idx / D) * D + idx % D];
  if (lane < 16) { xs[lane][D] = 1.0f; xs[lane][D + 1] = 0.0f; }

  for (int fi = lane; fi < KF; fi += 32) {
    int i, j;
    if (fi < D) { i = fi; j = D; }
    else if (fi < D + NPAIR) {
      int rem = fi - D, cnt = D; i = 0;
      while (rem >= cnt) { rem -= cnt; --cnt; ++i; }
      j = i + rem;
    } else if (fi == D + NPAIR) { i = D; j = D; }
    else { i = D + 1; j = D + 1; }
    lut[fi] = (unsigned short)((i << 8) | j);
  }
  __syncthreads();

  const v16bf* __restrict__ Wv = (const v16bf*)Wf;
  v8f acc[NT];
#pragma unroll
  for (int n = 0; n < NT; ++n)
#pragma unroll
    for (int r = 0; r < 8; ++r) acc[n][r] = 0.0f;

  for (int c = 0; c < NCH; ++c) {
    v16bf afrag;
#pragma unroll
    for (int e = 0; e < 16; ++e) {
      int kk = e + 16 * g;
      unsigned p = lut[c * 32 + kk];
      afrag[e] = (__bf16)(xs[row][p >> 8] * xs[row][p & 255]);
    }
#pragma unroll
    for (int n = 0; n < NT; ++n) {
      v16bf bfrag = Wv[(size_t)(c * NT + n) * 32 + lane];
      acc[n] = __builtin_amdgcn_wmma_f32_16x16x32_bf16(
          false, afrag, false, bfrag, (short)0, acc[n], false, false);
    }
  }

#pragma unroll
  for (int n = 0; n < NT; ++n)
#pragma unroll
    for (int r = 0; r < 8; ++r) {
      long m = base + r + 8 * g;
      out[m * NK + n * 16 + (lane & 15)] = acc[n][r];
    }
}

// =========================================================
__global__ void softmax_rows_kernel(const float* __restrict__ in,
                                    float* __restrict__ out,
                                    int rows, int cols)
{
  int r = blockIdx.x * blockDim.x + threadIdx.x;
  if (r >= rows) return;
  const float* p = in + (size_t)r * cols;
  float* o = out + (size_t)r * cols;
  float m = -1e30f;
  for (int j = 0; j < cols; ++j) m = fmaxf(m, p[j]);
  float s = 0.0f;
  for (int j = 0; j < cols; ++j) s += expf(p[j] - m);
  float inv = 1.0f / s;
  for (int j = 0; j < cols; ++j) o[j] = expf(p[j] - m) * inv;
}

// =========================================================
// Planning kernel: R[A,S], soft value iteration, Poisson mixing -> Q[A,S].
// =========================================================
__global__ __launch_bounds__(256)
void plan_kernel(const float* __restrict__ Bp,
                 const float* __restrict__ obs_lv,
                 const float* __restrict__ C_logits,
                 const float* __restrict__ tau,
                 float* __restrict__ Qout)
{
  __shared__ float R[NA][NS], Qc[NA][NS], accQ[NA][NS], V[NS], hw[HH];
  __shared__ float obs_ent[NS], sumlogC;
  const int t = threadIdx.x;

  if (t < NS) {
    float s = 0.0f;
    for (int j = 0; j < DOBS; ++j) s += obs_lv[t * DOBS + j];
    obs_ent[t] = 0.5f * (float)DOBS * (1.0f + logf(TWO_PI_F)) + s;
  }
  if (t == 0) {
    float m = -1e30f;
    for (int j = 0; j < NS; ++j) m = fmaxf(m, C_logits[j]);
    float se = 0.0f;
    for (int j = 0; j < NS; ++j) se += expf(C_logits[j] - m);
    float lse = m + logf(se);
    float s = 0.0f;
    for (int j = 0; j < NS; ++j) s += C_logits[j] - lse;
    sumlogC = s;
    float rate = expf(tau[0]);
    float lr = logf(rate), lf = 0.0f, mh = -1e30f;
    for (int k = 0; k < HH; ++k) {
      float lh = (float)k * lr - rate - lf;
      hw[k] = lh; mh = fmaxf(mh, lh);
      lf += logf((float)(k + 1));
    }
    float sh = 0.0f;
    for (int k = 0; k < HH; ++k) { hw[k] = expf(hw[k] - mh); sh += hw[k]; }
    for (int k = 0; k < HH; ++k) hw[k] /= sh;
  }
  __syncthreads();

  for (int idx = t; idx < NA * NS; idx += 256) {
    int a = idx >> 6, s2 = idx & 63;
    const float* row = Bp + ((size_t)a * NS + s2) * NS;
    float kl = 0.0f;
    for (int j = 0; j < NS; ++j) { float p = row[j]; kl += p * logf(p); }
    kl -= sumlogC;
    float r = -kl - obs_ent[s2];
    R[a][s2] = r; Qc[a][s2] = r; accQ[a][s2] = hw[0] * r;
  }
  __syncthreads();

  for (int it = 1; it < HH; ++it) {
    if (t < NS) {
      float m = -1e30f;
      for (int a = 0; a < NA; ++a) m = fmaxf(m, Qc[a][t]);
      float s = 0.0f;
      for (int a = 0; a < NA; ++a) s += expf(Qc[a][t] - m);
      V[t] = m + logf(s);
    }
    __syncthreads();
    for (int idx = t; idx < NA * NS; idx += 256) {
      int a = idx >> 6, s2 = idx & 63;
      const float* row = Bp + ((size_t)a * NS + s2) * NS;
      float acc2 = 0.0f;
      for (int j = 0; j < NS; ++j) acc2 += row[j] * V[j];
      float qn = R[a][s2] + acc2;
      Qc[a][s2] = qn; accQ[a][s2] += hw[it] * qn;
    }
    __syncthreads();
  }
  for (int idx = t; idx < NA * NS; idx += 256) Qout[idx] = accQ[idx >> 6][idx & 63];
}

// =========================================================
// Fused HMM scan + EFE outputs on matrix cores.
// 16 batches per block (4 blocks). Per step:
//   tmp[n,(a,j)] = sum_s b[n,s]*Bp[a,s,j]  -> [16x64]@[64x1024] GEMM,
//   64 N-tiles x 2 K-chunks of v_wmma_f32_16x16x32_bf16 (8 tiles/wave),
//   Bp LDS-resident as pre-packed bf16 B-fragments (128KB of 320KB WGP LDS).
//   G[n,a] = b . Q^T  -> 2 more WMMAs (wave 0).
// Per-step input rows are register double-buffered: step t+1's
// logp_o / logp_u / p_a rows are loaded into VGPRs right after the
// top-of-step barrier and retire behind the WMMA phase, removing L2
// latency from the 500-step sequential critical path.
// Epilogue: 16 lane-teams of 16, shfl_xor tree reductions, f32 renorm.
// =========================================================
__global__ __launch_bounds__(256)
void scan_wmma_kernel(const float* __restrict__ Bp,  const float* __restrict__ Qg,
                      const float* __restrict__ b0g, const float* __restrict__ lpo,
                      const float* __restrict__ lpu, const float* __restrict__ pag,
                      float* __restrict__ out_pi,    float* __restrict__ out_obs)
{
  __shared__ __align__(32) __bf16 Bfrag[2 * 64 * 32 * 16];  // 131072 B
  __shared__ __align__(32) __bf16 Qfrag[2 * 32 * 16];       // 2048 B
  __shared__ float bs[16][NS];          // beliefs (f32)
  __shared__ float tmp[16 * 1024];      // 65536 B
  __shared__ float prior[16][NS];
  __shared__ __align__(16) float lo[16 * NS];
  __shared__ float lu[16 * NA], pav[16 * NA], Gs[16 * NA];
  const int t = threadIdx.x;
  const int lane = t & 31, w = t >> 5;
  const int n0 = blockIdx.x * 16;

  // Pack Bp into bf16 WMMA B-fragments: composite col c2 = a*64+j, K = s.
  for (int idx = t; idx < 2 * 64 * 32 * 16; idx += 256) {
    int e = idx & 15, l = (idx >> 4) & 31, nt = (idx >> 9) & 63, c = idx >> 15;
    int col = 16 * nt + (l & 15);
    int K = 32 * c + e + 16 * (l >> 4);
    int a = col >> 6, j = col & 63;
    Bfrag[idx] = (__bf16)Bp[((size_t)a * NS + K) * NS + j];
  }
  // Q^T fragments: col = a, K = s.
  for (int idx = t; idx < 2 * 32 * 16; idx += 256) {
    int e = idx & 15, l = (idx >> 4) & 31, c = idx >> 9;
    int a = l & 15;
    int K = 32 * c + e + 16 * (l >> 4);
    Qfrag[idx] = (__bf16)Qg[a * NS + K];
  }
  for (int idx = t; idx < 16 * NS; idx += 256) bs[idx >> 6][idx & 63] = b0g[idx & 63];

  const v16bf* __restrict__ Bv = (const v16bf*)Bfrag;
  const v16bf* __restrict__ Qv = (const v16bf*)Qfrag;

  // prefetch step-0 rows into registers (4 lo values + 1 lu + 1 pa per thread)
  float4 r_lo = ((const float4*)(lpo + (size_t)n0 * NS))[t];
  float  r_lu = lpu[(size_t)n0 * NA + t];
  float  r_pa = pag[(size_t)n0 * NA + t];
  __syncthreads();

  for (int tt = 0; tt < TT; ++tt) {
    const size_t rowbase = (size_t)tt * NB + n0;
    // commit this step's prefetched rows to LDS
    *(float4*)&lo[t * 4] = r_lo;
    lu[t]  = r_lu;
    pav[t] = r_pa;
    __syncthreads();

    // issue next step's row loads; they retire behind the WMMA phase
    if (tt + 1 < TT) {
      const size_t rb2 = rowbase + NB;
      r_lo = ((const float4*)(lpo + rb2 * NS))[t];
      r_lu = lpu[rb2 * NA + t];
      r_pa = pag[rb2 * NA + t];
    }

    // A fragments from current beliefs (rows = batches), identical per wave.
    v16bf a0, a1;
    {
      const int row = lane & 15, g = lane >> 4;
#pragma unroll
      for (int e = 0; e < 16; ++e) {
        int kk = e + 16 * g;
        a0[e] = (__bf16)bs[row][kk];
        a1[e] = (__bf16)bs[row][32 + kk];
      }
    }
#pragma unroll
    for (int qd = 0; qd < 8; ++qd) {
      const int nt = w * 8 + qd;
      v8f acc;
#pragma unroll
      for (int r = 0; r < 8; ++r) acc[r] = 0.0f;
      acc = __builtin_amdgcn_wmma_f32_16x16x32_bf16(
          false, a0, false, Bv[(0 * 64 + nt) * 32 + lane], (short)0, acc, false, false);
      acc = __builtin_amdgcn_wmma_f32_16x16x32_bf16(
          false, a1, false, Bv[(1 * 64 + nt) * 32 + lane], (short)0, acc, false, false);
#pragma unroll
      for (int r = 0; r < 8; ++r)
        tmp[(r + 8 * (lane >> 4)) * 1024 + nt * 16 + (lane & 15)] = acc[r];
    }
    if (w == 0) {  // G = b x Q^T (pre-update beliefs)
      v8f acc;
#pragma unroll
      for (int r = 0; r < 8; ++r) acc[r] = 0.0f;
      acc = __builtin_amdgcn_wmma_f32_16x16x32_bf16(
          false, a0, false, Qv[0 * 32 + lane], (short)0, acc, false, false);
      acc = __builtin_amdgcn_wmma_f32_16x16x32_bf16(
          false, a1, false, Qv[1 * 32 + lane], (short)0, acc, false, false);
#pragma unroll
      for (int r = 0; r < 8; ++r)
        Gs[(r + 8 * (lane >> 4)) * 16 + (lane & 15)] = acc[r];
    }
    __syncthreads();

    // prior[n][j] = sum_a pav[n][a] * tmp[n][a*64+j]
    for (int idx = t; idx < 1024; idx += 256) {
      int nn = idx >> 6, j = idx & 63;
      const float* tr = &tmp[nn * 1024 + j];
      float s = 0.0f;
#pragma unroll
      for (int a = 0; a < NA; ++a) s += pav[nn * 16 + a] * tr[a * 64];
      prior[nn][j] = s;
    }
    __syncthreads();

    // per-batch epilogue: 16 teams of 16 lanes (teams stay inside a wave)
    {
      const int nn = t >> 4, l16 = t & 15;
      const int s0 = l16 * 4;
      float lov[4], pv[4];
      float m = -1e30f;
#pragma unroll
      for (int r = 0; r < 4; ++r) { lov[r] = lo[nn * 64 + s0 + r]; m = fmaxf(m, lov[r]); }
#pragma unroll
      for (int msk = 1; msk < 16; msk <<= 1) m = fmaxf(m, __shfl_xor(m, msk, 16));
      float ssum = 0.0f;
#pragma unroll
      for (int r = 0; r < 4; ++r) { pv[r] = __expf(lov[r] - m) * prior[nn][s0 + r]; ssum += pv[r]; }
#pragma unroll
      for (int msk = 1; msk < 16; msk <<= 1) ssum += __shfl_xor(ssum, msk, 16);
      const float inv = 1.0f / ssum;
      float vv[4], vmax = -1e30f;
#pragma unroll
      for (int r = 0; r < 4; ++r) {
        float bn = pv[r] * inv;
        bs[nn][s0 + r] = bn;
        vv[r] = __logf(bn + 1e-6f) * lov[r];
        vmax = fmaxf(vmax, vv[r]);
      }
#pragma unroll
      for (int msk = 1; msk < 16; msk <<= 1) vmax = fmaxf(vmax, __shfl_xor(vmax, msk, 16));
      float vs = 0.0f;
#pragma unroll
      for (int r = 0; r < 4; ++r) vs += __expf(vv[r] - vmax);
#pragma unroll
      for (int msk = 1; msk < 16; msk <<= 1) vs += __shfl_xor(vs, msk, 16);
      if (l16 == 0) out_obs[rowbase + nn] = vmax + __logf(vs);

      // logp_pi = lse_a(log_softmax(G) + lu)
      float gval = Gs[nn * 16 + l16];
      float m1 = gval;
#pragma unroll
      for (int msk = 1; msk < 16; msk <<= 1) m1 = fmaxf(m1, __shfl_xor(m1, msk, 16));
      float s1 = __expf(gval - m1);
#pragma unroll
      for (int msk = 1; msk < 16; msk <<= 1) s1 += __shfl_xor(s1, msk, 16);
      float lse1 = m1 + __logf(s1);
      float val = gval - lse1 + lu[nn * 16 + l16];
      float m2 = val;
#pragma unroll
      for (int msk = 1; msk < 16; msk <<= 1) m2 = fmaxf(m2, __shfl_xor(m2, msk, 16));
      float s2 = __expf(val - m2);
#pragma unroll
      for (int msk = 1; msk < 16; msk <<= 1) s2 += __shfl_xor(s2, msk, 16);
      if (l16 == 0) out_pi[rowbase + nn] = m2 + __logf(s2);
    }
    __syncthreads();
  }
}

// =========================================================
extern "C" void kernel_launch(void* const* d_in, const int* in_sizes, int n_in,
                              void* d_out, int out_size, void* d_ws, size_t ws_size,
                              hipStream_t stream)
{
  (void)in_sizes; (void)n_in; (void)out_size; (void)ws_size;
  const float* o        = (const float*)d_in[0];
  const float* u        = (const float*)d_in[1];
  const float* obs_mu   = (const float*)d_in[2];
  const float* obs_lv   = (const float*)d_in[3];
  const float* obs_tl   = (const float*)d_in[4];
  const float* ctl_mu   = (const float*)d_in[5];
  const float* ctl_lv   = (const float*)d_in[6];
  const float* ctl_tl   = (const float*)d_in[7];
  const float* B_logits = (const float*)d_in[8];
  const float* D_logits = (const float*)d_in[9];
  const float* C_logits = (const float*)d_in[10];
  const float* tau      = (const float*)d_in[11];

  char* ws = (char*)d_ws;
  float*  Bp   = (float*)(ws + OFF_BP);
  float*  Q    = (float*)(ws + OFF_Q);
  float*  b0   = (float*)(ws + OFF_B0);
  __bf16* Wobs = (__bf16*)(ws + OFF_WOBS);
  __bf16* Wctl = (__bf16*)(ws + OFF_WCTL);
  float*  lpo  = (float*)(ws + OFF_LPO);
  float*  lpu  = (float*)(ws + OFF_LPU);
  float*  pa   = (float*)(ws + OFF_PA);

  // setup: precision-form MVN params -> WMMA-B-fragment weights
  mvn_params_kernel<DOBS, 4, 18><<<NS, 32, 0, stream>>>(obs_mu, obs_lv, obs_tl, Wobs);
  mvn_params_kernel<DCTL, 1, 2 ><<<NA, 32, 0, stream>>>(ctl_mu, ctl_lv, ctl_tl, Wctl);
  softmax_rows_kernel<<<4, 256, 0, stream>>>(B_logits, Bp, NA * NS, NS);
  softmax_rows_kernel<<<1, 64, 0, stream>>>(D_logits, b0, 1, NS);
  plan_kernel<<<1, 256, 0, stream>>>(Bp, obs_lv, C_logits, tau, Q);

  // big WMMA GEMMs: logp_o [32000,64], logp_u [32000,16]
  gemm_logp_kernel<DOBS, 18, 4, NS><<<TB / 16, 32, 0, stream>>>(o, Wobs, lpo);
  gemm_logp_kernel<DCTL, 2, 1, NA><<<TB / 16, 32, 0, stream>>>(u, Wctl, lpu);
  softmax_rows_kernel<<<125, 256, 0, stream>>>(lpu, pa, TB, NA);

  // fused WMMA HMM scan + outputs (16 batches per block)
  scan_wmma_kernel<<<NB / 16, 256, 0, stream>>>(Bp, Q, b0, lpo, lpu, pa,
                                                (float*)d_out, (float*)d_out + TB);
}
